// TransformerBlock_25786983645956
// MI455X (gfx1250) — matrix operand order repaired and checked by reading, not compiled, not run on hardware
//
#include <hip/hip_runtime.h>
#include <hip/hip_bf16.h>
#include <math.h>

typedef __bf16 bf16;
typedef __attribute__((ext_vector_type(16))) __bf16 v16bf;
typedef __attribute__((ext_vector_type(8)))  __bf16 v8bf;
typedef __attribute__((ext_vector_type(8)))  float  v8f;

union V16 { v16bf v; v8bf h[2]; };

constexpr int Bc = 2, Sc = 2048, Dc = 1024, Hc = 16, DHc = 64, DMc = 4096;
constexpr float EPSc = 1e-5f;
constexpr float IGNOREc = -100000.0f;

// ---------------------------------------------------------------- utilities

__global__ void __launch_bounds__(256) cvt_bf16_kernel(const float* __restrict__ in,
                                                       bf16* __restrict__ out, int n) {
    int i = blockIdx.x * 256 + threadIdx.x;
    if (i < n) out[i] = (bf16)in[i];
}

// W_O is (H, D, DH); permute to wmat[d][i*DH+h] = W_O[i][d][h]
__global__ void __launch_bounds__(256) cvt_wo_kernel(const float* __restrict__ wo,
                                                     bf16* __restrict__ out) {
    int idx = blockIdx.x * 256 + threadIdx.x;   // over H*D*DH = 1<<20
    int h = idx & 63;
    int d = (idx >> 6) & (Dc - 1);
    int i = idx >> 16;
    out[(size_t)d * (Hc * DHc) + i * DHc + h] = (bf16)wo[idx];
}

__device__ inline float block_sum256(float v, float* sm) {
    #pragma unroll
    for (int m = 16; m; m >>= 1) v += __shfl_xor(v, m);
    int lane = threadIdx.x & 31, w = threadIdx.x >> 5;
    if (lane == 0) sm[w] = v;
    __syncthreads();
    float tot = 0.f;
    #pragma unroll
    for (int i = 0; i < 8; ++i) tot += sm[i];
    __syncthreads();
    return tot;
}

__global__ void __launch_bounds__(256) ln_bf16_kernel(const float* __restrict__ x,
                                                      bf16* __restrict__ out, int D) {
    __shared__ float sm[8];
    int row = blockIdx.x;
    const float* xr = x + (size_t)row * D;
    int t = threadIdx.x;
    float s = 0.f;
    for (int i = t; i < D; i += 256) s += xr[i];
    float mean = block_sum256(s, sm) / (float)D;
    float v = 0.f;
    for (int i = t; i < D; i += 256) { float d = xr[i] - mean; v += d * d; }
    float var = block_sum256(v, sm) / (float)D;
    float inv = rsqrtf(var + EPSc);
    bf16* o = out + (size_t)row * D;
    for (int i = t; i < D; i += 256) o[i] = (bf16)((xr[i] - mean) * inv);
}

__device__ inline float gelu_new_f(float x) {
    const float c = 0.7978845608028654f;  // sqrt(2/pi)
    return 0.5f * x * (1.0f + tanhf(c * (x + 0.044715f * x * x * x)));
}

// ---------------------------------------------------------------- WMMA GEMM
// Y[M,N] = A[M,K](bf16, row-major) x W[N,K](bf16, row-major)^T + bias
// Block tile 128(M) x 64(N), BK=32, double-buffered LDS, 4 WMMA / wave / K-step.

enum { OUT_QK = 0, OUT_V = 1, OUT_F32_RESID = 2, OUT_BF16_GELU = 3 };

template <int MODE>
__global__ void __launch_bounds__(256) wmma_gemm_kernel(
    const bf16* __restrict__ A, const bf16* __restrict__ W,
    const float* __restrict__ bias, const float* __restrict__ resid,
    void* __restrict__ outp, int M, int N, int K) {
    __shared__ __align__(16) bf16 lsA[2][128][32];
    __shared__ __align__(16) bf16 lsB[2][64][32];

    int t = threadIdx.x;
    int lane = t & 31, wave = t >> 5;
    int half = lane >> 4, ln = lane & 15;
    int wm = (wave & 3) * 32;       // wave's 32-row slice of the block tile
    int wn = (wave >> 2) * 32;      // wave's 32-col slice
    int row0 = blockIdx.y * 128;
    int col0 = blockIdx.x * 64;

    // staging decomposition (256 threads)
    int ar = t >> 1, ac = (t & 1) * 16;   // A: 128 rows x 32, 16 bf16/thread
    int br = t >> 2, bc = (t & 3) * 8;    // B: 64 rows x 32,   8 bf16/thread

    const bf16* aptr = &A[(size_t)(row0 + ar) * K + ac];
    const bf16* wptr = &W[(size_t)(col0 + br) * K + bc];

    auto stage = [&](int buf, int k0) {
        *(v8bf*)&lsA[buf][ar][ac]     = *(const v8bf*)(aptr + k0);
        *(v8bf*)&lsA[buf][ar][ac + 8] = *(const v8bf*)(aptr + k0 + 8);
        *(v8bf*)&lsB[buf][br][bc]     = *(const v8bf*)(wptr + k0);
    };

    v8f acc[2][2] = {};

    int nk = K >> 5;
    stage(0, 0);
    __syncthreads();

    for (int it = 0; it < nk; ++it) {
        int cur = it & 1;
        if (it + 1 < nk) {
            stage(cur ^ 1, (it + 1) * 32);
            if (it + 2 < nk) {  // pull HBM lines toward L2 early (global_prefetch_b8)
                __builtin_prefetch(aptr + (it + 2) * 32, 0, 0);
                __builtin_prefetch(wptr + (it + 2) * 32, 0, 0);
            }
        }
        // A fragments (16x32): lane half holds K {half*8..+7, 16+half*8..+7}
        V16 a0, a1;
        a0.h[0] = *(const v8bf*)&lsA[cur][wm + ln][half * 8];
        a0.h[1] = *(const v8bf*)&lsA[cur][wm + ln][16 + half * 8];
        a1.h[0] = *(const v8bf*)&lsA[cur][wm + 16 + ln][half * 8];
        a1.h[1] = *(const v8bf*)&lsA[cur][wm + 16 + ln][16 + half * 8];
        // B fragments (32x16): lane half holds K {half*8..+7, 16+half*8..+7}
        V16 b0, b1;
        b0.h[0] = *(const v8bf*)&lsB[cur][wn + ln][half * 8];
        b0.h[1] = *(const v8bf*)&lsB[cur][wn + ln][16 + half * 8];
        b1.h[0] = *(const v8bf*)&lsB[cur][wn + 16 + ln][half * 8];
        b1.h[1] = *(const v8bf*)&lsB[cur][wn + 16 + ln][16 + half * 8];

        acc[0][0] = __builtin_amdgcn_wmma_f32_16x16x32_bf16(false, a0.v, false, b0.v,
                                                            (short)0, acc[0][0], false, false);
        acc[0][1] = __builtin_amdgcn_wmma_f32_16x16x32_bf16(false, a0.v, false, b1.v,
                                                            (short)0, acc[0][1], false, false);
        acc[1][0] = __builtin_amdgcn_wmma_f32_16x16x32_bf16(false, a1.v, false, b0.v,
                                                            (short)0, acc[1][0], false, false);
        acc[1][1] = __builtin_amdgcn_wmma_f32_16x16x32_bf16(false, a1.v, false, b1.v,
                                                            (short)0, acc[1][1], false, false);
        __syncthreads();
    }

    #pragma unroll
    for (int i = 0; i < 2; ++i) {
        #pragma unroll
        for (int j = 0; j < 2; ++j) {
            int nn = col0 + wn + j * 16 + ln;
            float bi = bias[nn];
            #pragma unroll
            for (int jj = 0; jj < 8; ++jj) {
                int m = row0 + wm + i * 16 + half * 8 + jj;
                float v = acc[i][j][jj] + bi;
                if constexpr (MODE == OUT_QK) {
                    int b = m >> 11, s = m & (Sc - 1);
                    int h = nn >> 6, dh = nn & 63;
                    ((bf16*)outp)[(((size_t)(b * Hc + h)) * Sc + s) * DHc + dh] = (bf16)v;
                } else if constexpr (MODE == OUT_V) {
                    int b = m >> 11, s = m & (Sc - 1);
                    int h = nn >> 6, dh = nn & 63;
                    ((bf16*)outp)[(((size_t)(b * Hc + h)) * DHc + dh) * Sc + s] = (bf16)v;
                } else if constexpr (MODE == OUT_F32_RESID) {
                    ((float*)outp)[(size_t)m * N + nn] = resid[(size_t)m * N + nn] + v;
                } else {  // OUT_BF16_GELU
                    ((bf16*)outp)[(size_t)m * N + nn] = (bf16)gelu_new_f(v);
                }
            }
        }
    }
}

// ---------------------------------------------------------------- Flash attention
// q,k: [B,H,S,DH] bf16 ; v: [B,H,DH,S] bf16 ; z out: [B,S,H*DH] bf16
// block = 4 waves (128 threads); wave = 16 query rows, full DH=64.

__global__ void __launch_bounds__(128) attn_kernel(const bf16* __restrict__ q,
                                                   const bf16* __restrict__ k,
                                                   const bf16* __restrict__ v,
                                                   bf16* __restrict__ z) {
    __shared__ __align__(16) bf16 lsP[4][16][32];

    int tid = threadIdx.x;
    int lane = tid & 31, wave = tid >> 5;
    int half = lane >> 4, ln = lane & 15;
    int bh = blockIdx.y;                 // b*H + h
    int b = bh >> 4, h = bh & (Hc - 1);
    int q0 = blockIdx.x * 64 + wave * 16;

    const bf16* qb = q + (size_t)bh * Sc * DHc;
    const bf16* kb = k + (size_t)bh * Sc * DHc;
    const bf16* vb = v + (size_t)bh * DHc * Sc;

    // Q fragments for dh chunks c=0 (0..31) and c=1 (32..63); resident all loop
    V16 af[2];
    #pragma unroll
    for (int c = 0; c < 2; ++c) {
        const bf16* qp = &qb[(size_t)(q0 + ln) * DHc + c * 32];
        af[c].h[0] = *(const v8bf*)&qp[half * 8];
        af[c].h[1] = *(const v8bf*)&qp[16 + half * 8];
    }

    v8f acc[4] = {};                     // O accumulators: dh tiles d=0..3
    float mrun[8], lrun[8];
    #pragma unroll
    for (int j = 0; j < 8; ++j) { mrun[j] = -1e30f; lrun[j] = 0.f; }

    int qhi = q0 + 15;
    for (int k0 = 0; k0 <= qhi; k0 += 32) {
        // ---- scores: two 16x16 tiles (keys k0.. and k0+16..), K-dim = DH = 2x32
        v8f sc[2];
        #pragma unroll
        for (int tt = 0; tt < 2; ++tt) {
            v8f s = {};
            #pragma unroll
            for (int c = 0; c < 2; ++c) {
                V16 bfr;
                const bf16* kp = &kb[(size_t)(k0 + tt * 16 + ln) * DHc + c * 32 + half * 8];
                bfr.h[0] = *(const v8bf*)kp;
                bfr.h[1] = *(const v8bf*)(kp + 16);
                s = __builtin_amdgcn_wmma_f32_16x16x32_bf16(false, af[c].v, false, bfr.v,
                                                            (short)0, s, false, false);
            }
            sc[tt] = s;
        }
        // ---- scale + causal mask
        #pragma unroll
        for (int tt = 0; tt < 2; ++tt) {
            int key = k0 + tt * 16 + ln;
            #pragma unroll
            for (int j = 0; j < 8; ++j) {
                int qr = q0 + half * 8 + j;
                float val = sc[tt][j] * 0.125f;   // 1/sqrt(DH)
                sc[tt][j] = (key <= qr) ? val : IGNOREc;
            }
        }
        // ---- online softmax (row lives across 16 lanes, one VGPR)
        #pragma unroll
        for (int j = 0; j < 8; ++j) {
            float mx = fmaxf(sc[0][j], sc[1][j]);
            #pragma unroll
            for (int msk = 8; msk; msk >>= 1) mx = fmaxf(mx, __shfl_xor(mx, msk));
            float nm = fmaxf(mrun[j], mx);
            float p0 = __expf(sc[0][j] - nm);
            float p1 = __expf(sc[1][j] - nm);
            float rs = p0 + p1;
            #pragma unroll
            for (int msk = 8; msk; msk >>= 1) rs += __shfl_xor(rs, msk);
            float scale = __expf(mrun[j] - nm);
            lrun[j] = lrun[j] * scale + rs;
            mrun[j] = nm;
            #pragma unroll
            for (int d = 0; d < 4; ++d) acc[d][j] *= scale;
            // stash P (acc layout) into per-wave LDS for A-layout reload
            lsP[wave][half * 8 + j][ln]      = (bf16)p0;
            lsP[wave][half * 8 + j][16 + ln] = (bf16)p1;
        }
        // intra-wave LDS RAW across lanes: drain DS before cross-lane reload
        __asm__ volatile("s_wait_dscnt 0" ::: "memory");

        // ---- O += P(16x32) x V(32x64): V fragments contiguous ([dh][key] layout)
        V16 pf;
        pf.h[0] = *(const v8bf*)&lsP[wave][ln][half * 8];
        pf.h[1] = *(const v8bf*)&lsP[wave][ln][16 + half * 8];
        #pragma unroll
        for (int d = 0; d < 4; ++d) {
            V16 vf;
            const bf16* vp = &vb[(size_t)(d * 16 + ln) * Sc + k0 + half * 8];
            vf.h[0] = *(const v8bf*)vp;
            vf.h[1] = *(const v8bf*)(vp + 16);
            acc[d] = __builtin_amdgcn_wmma_f32_16x16x32_bf16(false, pf.v, false, vf.v,
                                                             (short)0, acc[d], false, false);
        }
        __asm__ volatile("s_wait_dscnt 0" ::: "memory");
    }

    // ---- epilogue: z[b, q, h*64+dh] = acc / l
    #pragma unroll
    for (int d = 0; d < 4; ++d) {
        int dh = d * 16 + ln;
        #pragma unroll
        for (int j = 0; j < 8; ++j) {
            int qr = q0 + half * 8 + j;
            float ov = acc[d][j] / lrun[j];
            z[((size_t)b * Sc + qr) * (Hc * DHc) + h * DHc + dh] = (bf16)ov;
        }
    }
}

// ---------------------------------------------------------------- launch

extern "C" void kernel_launch(void* const* d_in, const int* in_sizes, int n_in,
                              void* d_out, int out_size, void* d_ws, size_t ws_size,
                              hipStream_t stream) {
    const float* x     = (const float*)d_in[0];
    const float* W_Q   = (const float*)d_in[1];
    const float* W_K   = (const float*)d_in[2];
    const float* W_V   = (const float*)d_in[3];
    const float* W_O   = (const float*)d_in[4];
    const float* b_Q   = (const float*)d_in[5];
    const float* b_K   = (const float*)d_in[6];
    const float* b_V   = (const float*)d_in[7];
    const float* b_O   = (const float*)d_in[8];
    const float* W_in  = (const float*)d_in[9];
    const float* b_in  = (const float*)d_in[10];
    const float* W_out = (const float*)d_in[11];
    const float* b_out = (const float*)d_in[12];

    const int M  = Bc * Sc;        // 4096
    const int NW = Hc * DHc;       // 1024 == D
    char* ws = (char*)d_ws;
    size_t off = 0;
    auto carve = [&](size_t bytes) { char* p = ws + off; off += (bytes + 255) & ~size_t(255); return p; };

    bf16* wq   = (bf16*)carve((size_t)NW * Dc * 2);
    bf16* wk   = (bf16*)carve((size_t)NW * Dc * 2);
    bf16* wv   = (bf16*)carve((size_t)NW * Dc * 2);
    bf16* wo   = (bf16*)carve((size_t)Dc * NW * 2);
    bf16* win  = (bf16*)carve((size_t)DMc * Dc * 2);
    bf16* wout = (bf16*)carve((size_t)Dc * DMc * 2);
    bf16* xn   = (bf16*)carve((size_t)M * Dc * 2);
    bf16* qbf  = (bf16*)carve((size_t)M * NW * 2);
    bf16* kbf  = (bf16*)carve((size_t)M * NW * 2);
    bf16* vbf  = (bf16*)carve((size_t)M * NW * 2);
    bf16* zbf  = (bf16*)carve((size_t)M * NW * 2);
    float* rmid = (float*)carve((size_t)M * Dc * 4);
    bf16* mbf  = (bf16*)carve((size_t)M * Dc * 2);
    bf16* post = (bf16*)carve((size_t)M * DMc * 2);

    // weight conversion
    int nQ = NW * Dc;              // 1,048,576
    cvt_bf16_kernel<<<(nQ + 255) / 256, 256, 0, stream>>>(W_Q, wq, nQ);
    cvt_bf16_kernel<<<(nQ + 255) / 256, 256, 0, stream>>>(W_K, wk, nQ);
    cvt_bf16_kernel<<<(nQ + 255) / 256, 256, 0, stream>>>(W_V, wv, nQ);
    cvt_wo_kernel<<<nQ / 256, 256, 0, stream>>>(W_O, wo);
    int nM = DMc * Dc;             // 4,194,304
    cvt_bf16_kernel<<<(nM + 255) / 256, 256, 0, stream>>>(W_in,  win,  nM);
    cvt_bf16_kernel<<<(nM + 255) / 256, 256, 0, stream>>>(W_out, wout, nM);

    // LN1
    ln_bf16_kernel<<<M, 256, 0, stream>>>(x, xn, Dc);

    // QKV projections  (grid: N/64 x M/128)
    dim3 gD(NW / 64, M / 128);     // (16,32)
    wmma_gemm_kernel<OUT_QK><<<gD, 256, 0, stream>>>(xn, wq, b_Q, nullptr, qbf, M, NW, Dc);
    wmma_gemm_kernel<OUT_QK><<<gD, 256, 0, stream>>>(xn, wk, b_K, nullptr, kbf, M, NW, Dc);
    wmma_gemm_kernel<OUT_V> <<<gD, 256, 0, stream>>>(xn, wv, b_V, nullptr, vbf, M, NW, Dc);

    // fused causal flash attention
    attn_kernel<<<dim3(Sc / 64, Bc * Hc), 128, 0, stream>>>(qbf, kbf, vbf, zbf);

    // O projection + residual (resid_mid = x + attn_out)
    wmma_gemm_kernel<OUT_F32_RESID><<<gD, 256, 0, stream>>>(zbf, wo, b_O, x, rmid, M, Dc, Dc);

    // LN2
    ln_bf16_kernel<<<M, 256, 0, stream>>>(rmid, mbf, Dc);

    // MLP in + GELU
    dim3 gM(DMc / 64, M / 128);    // (64,32)
    wmma_gemm_kernel<OUT_BF16_GELU><<<gM, 256, 0, stream>>>(mbf, win, b_in, nullptr, post, M, DMc, Dc);

    // MLP out + residual -> d_out (f32)
    wmma_gemm_kernel<OUT_F32_RESID><<<gD, 256, 0, stream>>>(post, wout, b_out, rmid,
                                                            (float*)d_out, M, Dc, DMc);
    (void)in_sizes; (void)n_in; (void)out_size; (void)ws_size;
}
